// FloodBlock_30906584662494
// MI455X (gfx1250) — compile-verified, matrix operand order checked by reading, not run on hardware
//
#include <hip/hip_runtime.h>
#include <hip/hip_bf16.h>
#include <math.h>

typedef __attribute__((ext_vector_type(16))) _Float16 v16h;
typedef __attribute__((ext_vector_type(8)))  float    v8f;

#define GVP_EPS 1e-8f

__device__ __forceinline__ float sigmoidf_(float x) { return 1.0f / (1.0f + __expf(-x)); }

// monotonic float -> uint mapping for atomicMax-based segment max
__device__ __forceinline__ unsigned fenc(float f) {
    unsigned u = __float_as_uint(f);
    return (u & 0x80000000u) ? ~u : (u | 0x80000000u);
}
__device__ __forceinline__ float fdec(unsigned e) {
    unsigned b = (e & 0x80000000u) ? (e ^ 0x80000000u) : ~e;
    return __uint_as_float(b);
}

// ---------------------------------------------------------------------------
// Fused GVP kernel.
//   vh  = v_in @ wh          (per spatial dim)          [rows x 8 x 3]
//   vn  = ||vh||_i                                     [rows x 8]
//   s_o = act( concat(s_in, vn) @ ws + b )             [rows x 32]   <-- WMMA
//   vo  = vh @ wv ; vo *= sigmoid(||vo||) (if VSIG)    [rows x 8 x 3]
// EDGE mode: rows are edges; s_in/v_in gathered via src[], pre-scaled by the
// per-edge softmax attention, and outputs atomically scattered to dst[].
// 256 threads = 8 waves; each wave owns one 16-row tile.
// Row counts (N=30000, Et=510000) are multiples of 16, so every 16-row wave
// tile is wholly valid or wholly empty; tile validity is forced into an SGPR
// via readfirstlane so stores sit behind ONE scalar branch.
// Chunk-1 (vn / tail-of-ws) fragments are built branch-free with a per-lane
// float mask instead of per-element divergent selects.
// ---------------------------------------------------------------------------
template<int CIN, bool RELU, bool VSIG, bool EDGE, bool ACCUM>
__global__ __launch_bounds__(256)
void k_gvp(const float* __restrict__ s_in, const float* __restrict__ v_in,
           const float* __restrict__ wh,   const float* __restrict__ ws,
           const float* __restrict__ bias, const float* __restrict__ wvw,
           float* __restrict__ s_out, float* __restrict__ v_out, int M,
           const int* __restrict__ ei, int E, int Et,
           const float* __restrict__ exp_s, const float* __restrict__ exp_v,
           const float* __restrict__ z_s,   const float* __restrict__ z_v)
{
    __shared__ float ws_l[40][32];
    __shared__ float wh_l[CIN][8];
    __shared__ float wv_l[8][8];
    __shared__ float b_l[32];
    __shared__ float s_l [8][16][32];
    __shared__ float vh_l[8][16][24];
    __shared__ float vn_l[8][16][8];
    __shared__ float vo_l[8][16][24];
    __shared__ int   src_l[8][16];
    __shared__ int   dst_l[8][16];
    __shared__ float as_l[8][16];
    __shared__ float av_l[8][16];

    const int t    = threadIdx.x;
    const int wave = t >> 5;
    const int lane = t & 31;
    const int tile0 = (blockIdx.x * 8 + wave) * 16;
    // wave-uniform tile validity, forced scalar (tile0/M are uniform per wave)
    const bool tile_valid = __builtin_amdgcn_readfirstlane(tile0 < M ? 1 : 0) != 0;

    // --- stage weights (block-wide) ---
    for (int i = t; i < 40 * 32; i += 256) ws_l[i >> 5][i & 31] = ws[i];
    for (int i = t; i < CIN * 8; i += 256) wh_l[i / 8][i % 8]   = wh[i];
    for (int i = t; i < 64;      i += 256) wv_l[i >> 3][i & 7]  = wvw[i];
    if (t < 32) b_l[t] = bias[t];

    // --- edge meta: src/dst/attention per row ---
    if (EDGE) {
        if (lane < 16) {
            int e = tile0 + lane;
            int si = 0, di = 0; float as = 0.f, av = 0.f;
            if (e < Et) {
                si = (e < E) ? ei[e]     : (e - E);
                di = (e < E) ? ei[E + e] : (e - E);
                as = exp_s[e] / (z_s[di] + 1e-16f);
                av = exp_v[e] / (z_v[di] + 1e-16f);
            }
            src_l[wave][lane] = si; dst_l[wave][lane] = di;
            as_l[wave][lane]  = as; av_l[wave][lane]  = av;
        }
    }
    __syncthreads();

    // --- stage s tile (attention-scaled in edge mode) ---
    for (int idx = lane; idx < 16 * 32; idx += 32) {
        int r = idx >> 5, k = idx & 31;
        float val;
        if (EDGE) {
            val = as_l[wave][r] * s_in[src_l[wave][r] * 32 + k];
        } else {
            int row = tile0 + r; if (row >= M) row = M - 1;
            val = s_in[row * 32 + k];
        }
        s_l[wave][r][k] = val;
    }
    // --- vh = (scaled) v @ wh ---
    for (int idx = lane; idx < 16 * 24; idx += 32) {
        int r = idx / 24, rem = idx % 24, h = rem / 3, i = rem % 3;
        int row; float scale = 1.0f;
        if (EDGE) { row = src_l[wave][r]; scale = av_l[wave][r]; }
        else      { row = tile0 + r; if (row >= M) row = M - 1; }
        float acc = 0.f;
#pragma unroll
        for (int c = 0; c < CIN; ++c)
            acc += v_in[(row * CIN + c) * 3 + i] * wh_l[c][h];
        vh_l[wave][r][h * 3 + i] = acc * scale;
    }
    __syncthreads();

    // --- vn = ||vh|| over spatial dim ---
    for (int idx = lane; idx < 16 * 8; idx += 32) {
        int r = idx >> 3, h = idx & 7;
        float x = vh_l[wave][r][h * 3 + 0];
        float y = vh_l[wave][r][h * 3 + 1];
        float z = vh_l[wave][r][h * 3 + 2];
        vn_l[wave][r][h] = sqrtf(fmaxf(x * x + y * y + z * z, GVP_EPS));
    }
    __syncthreads();

    // --- WMMA: s_out(16x32) = [s | vn | 0pad](16x64) @ ws(64x32) ---
    // A fragment layout (16-bit A 16x32, wave32): lane m=L&15; lanes<16 hold
    // K = {0..7,16..23}, lanes>=16 hold K = {8..15,24..31}.
    // Chunk 1 (K=32..63) only has data at K=32..39 (vn / ws rows 32..39),
    // which lands exclusively in lanes<16 at hh<8 -> per-lane float mask.
    const int   m       = lane & 15;
    const int   ko      = (lane < 16) ? 0 : 8;
    const float lo_mask = (lane < 16) ? 1.0f : 0.0f;

    v16h a0, a1;
#pragma unroll
    for (int hh = 0; hh < 16; ++hh) {
        int kl = ko + (hh & 7) + ((hh & 8) ? 16 : 0);
        a0[hh] = (_Float16)s_l[wave][m][kl];                  // K chunk 0: s
    }
#pragma unroll
    for (int hh = 0; hh < 8; ++hh)
        a1[hh] = (_Float16)(vn_l[wave][m][hh] * lo_mask);     // K chunk 1: vn
#pragma unroll
    for (int hh = 8; hh < 16; ++hh)
        a1[hh] = (_Float16)0.0f;                              // zero pad

#pragma unroll
    for (int nt = 0; nt < 2; ++nt) {
        const int n = (lane & 15) + nt * 16;
        v16h b0, b1;
#pragma unroll
        for (int hh = 0; hh < 16; ++hh) {
            int kl = ko + (hh & 7) + ((hh & 8) ? 16 : 0);
            b0[hh] = (_Float16)ws_l[kl][n];
        }
#pragma unroll
        for (int hh = 0; hh < 8; ++hh)
            b1[hh] = (_Float16)(ws_l[32 + hh][n] * lo_mask);  // ws rows 32..39
#pragma unroll
        for (int hh = 8; hh < 16; ++hh)
            b1[hh] = (_Float16)0.0f;

        v8f acc = {};
        acc = __builtin_amdgcn_wmma_f32_16x16x32_f16(false, a0, false, b0,
                                                     (short)0, acc, false, false);
        acc = __builtin_amdgcn_wmma_f32_16x16x32_f16(false, a1, false, b1,
                                                     (short)0, acc, false, false);
        // epilogue: D layout -> lane n = L&15 (+16*nt); elem j -> row j+8*(L>=16)
        if (tile_valid) {
            const float bn   = b_l[n];
            const int   moff = (lane < 16) ? 0 : 8;
            if (EDGE) {
#pragma unroll
                for (int j = 0; j < 8; ++j) {
                    float r = acc[j] + bn;
                    if (RELU) r = fmaxf(r, 0.f);
                    atomicAdd(&s_out[dst_l[wave][j + moff] * 32 + n], r);
                }
            } else {
#pragma unroll
                for (int j = 0; j < 8; ++j) {
                    int row = tile0 + j + moff;
                    float r = acc[j] + bn;
                    if (RELU) r = fmaxf(r, 0.f);
                    if (ACCUM) s_out[row * 32 + n] += r;
                    else       s_out[row * 32 + n]  = r;
                }
            }
        }
    }

    // --- vo = vh @ wv ---
    for (int idx = lane; idx < 16 * 24; idx += 32) {
        int r = idx / 24, rem = idx % 24, c = rem / 3, i = rem % 3;
        float acc = 0.f;
#pragma unroll
        for (int h = 0; h < 8; ++h)
            acc += vh_l[wave][r][h * 3 + i] * wv_l[h][c];
        vo_l[wave][r][c * 3 + i] = acc;
    }
    __syncthreads();

    // --- vector gate + writeback (uniform tile_valid branch) ---
    if (tile_valid) {
        for (int idx = lane; idx < 16 * 8; idx += 32) {
            int r = idx >> 3, c = idx & 7;
            float x = vo_l[wave][r][c * 3 + 0];
            float y = vo_l[wave][r][c * 3 + 1];
            float z = vo_l[wave][r][c * 3 + 2];
            float nrm = sqrtf(fmaxf(x * x + y * y + z * z, GVP_EPS));
            float sc  = VSIG ? sigmoidf_(nrm) : 1.0f;
            int row = tile0 + r;
            if (EDGE) {
                int d = dst_l[wave][r];
                atomicAdd(&v_out[(d * 8 + c) * 3 + 0], x * sc);
                atomicAdd(&v_out[(d * 8 + c) * 3 + 1], y * sc);
                atomicAdd(&v_out[(d * 8 + c) * 3 + 2], z * sc);
            } else if (ACCUM) {
                v_out[(row * 8 + c) * 3 + 0] += x * sc;
                v_out[(row * 8 + c) * 3 + 1] += y * sc;
                v_out[(row * 8 + c) * 3 + 2] += z * sc;
            } else {
                v_out[(row * 8 + c) * 3 + 0] = x * sc;
                v_out[(row * 8 + c) * 3 + 1] = y * sc;
                v_out[(row * 8 + c) * 3 + 2] = z * sc;
            }
        }
    }
}

// --- attention pass 1: raw leaky-relu scores + segment max ---
__global__ void k_att_raw(const float* __restrict__ s_e, const float* __restrict__ v_e,
                          const int* __restrict__ ei, int E, int Et,
                          float* __restrict__ raw_s, float* __restrict__ raw_v,
                          unsigned* __restrict__ max_s, unsigned* __restrict__ max_v)
{
    int e = blockIdx.x * blockDim.x + threadIdx.x;
    if (e >= Et) return;
    int si = (e < E) ? ei[e]     : (e - E);
    int di = (e < E) ? ei[E + e] : (e - E);
    const float* a = s_e + (size_t)di * 32;
    const float* b = s_e + (size_t)si * 32;
    float ds = 0.f;
#pragma unroll
    for (int k = 0; k < 32; ++k) ds += a[k] * b[k];
    ds = (ds > 0.f) ? ds : 0.2f * ds;
    const float* c = v_e + (size_t)di * 24;
    const float* d = v_e + (size_t)si * 24;
    float dv = 0.f;
#pragma unroll
    for (int k = 0; k < 24; ++k) dv += c[k] * d[k];
    dv = (dv > 0.f) ? dv : 0.2f * dv;
    raw_s[e] = ds; raw_v[e] = dv;
    atomicMax(&max_s[di], fenc(ds));
    atomicMax(&max_v[di], fenc(dv));
}

// --- attention pass 2: exp(raw - max) + segment sum ---
__global__ void k_att_exp(const int* __restrict__ ei, int E, int Et,
                          const float* __restrict__ raw_s, const float* __restrict__ raw_v,
                          const unsigned* __restrict__ max_s, const unsigned* __restrict__ max_v,
                          float* __restrict__ exp_s, float* __restrict__ exp_v,
                          float* __restrict__ z_s, float* __restrict__ z_v)
{
    int e = blockIdx.x * blockDim.x + threadIdx.x;
    if (e >= Et) return;
    int di = (e < E) ? ei[E + e] : (e - E);
    float es = __expf(raw_s[e] - fdec(max_s[di]));
    float ev = __expf(raw_v[e] - fdec(max_v[di]));
    exp_s[e] = es; exp_v[e] = ev;
    atomicAdd(&z_s[di], es);
    atomicAdd(&z_v[di], ev);
}

__global__ void k_fill(float* __restrict__ p, float v, int n) {
    int i = blockIdx.x * blockDim.x + threadIdx.x;
    if (i < n) p[i] = v;
}

// s_rs = sigmoid(gr) * hid   (elementwise; used for both s and v parts)
__global__ void k_gate_scale(const float* __restrict__ gr, const float* __restrict__ hid,
                             float* __restrict__ out, int n)
{
    int i = blockIdx.x * blockDim.x + threadIdx.x;
    if (i < n) out[i] = sigmoidf_(gr[i]) * hid[i];
}

// GRU combine + mask, writes the concatenated (s_t, v_t) output.
__global__ void k_final(const float* __restrict__ s_gz, const float* __restrict__ s_hh,
                        const float* __restrict__ s_hid,
                        const float* __restrict__ v_gz, const float* __restrict__ v_hh,
                        const float* __restrict__ v_hid,
                        const unsigned char* __restrict__ mask,
                        float* __restrict__ out, int N)
{
    int i = blockIdx.x * blockDim.x + threadIdx.x;
    int ns = N * 32;
    int total = N * 56;
    if (i >= total) return;
    if (i < ns) {
        int node = i >> 5;
        float gz = sigmoidf_(s_gz[i]);
        float hh = tanhf(s_hh[i]);
        float st = gz * s_hid[i] + (1.f - gz) * hh;
        out[i] = mask[node] ? st : 0.f;
    } else {
        int iv = i - ns;
        int node = iv / 24;
        float gz = sigmoidf_(v_gz[iv]);
        float hh = tanhf(v_hh[iv]);
        float vt = gz * v_hid[iv] + (1.f - gz) * hh;
        out[ns + iv] = mask[node] ? vt : 0.f;
    }
}

extern "C" void kernel_launch(void* const* d_in, const int* in_sizes, int n_in,
                              void* d_out, int out_size, void* d_ws, size_t ws_size,
                              hipStream_t stream)
{
    const int*   ei      = (const int*)d_in[0];
    const float* s_raw   = (const float*)d_in[1];
    const float* v_raw   = (const float*)d_in[2];
    const float* s_hid   = (const float*)d_in[3];
    const float* v_hid   = (const float*)d_in[4];
    const unsigned char* mask = (const unsigned char*)d_in[5];   // numpy bool
    const float* g0_wh   = (const float*)d_in[6];
    const float* g0_ws   = (const float*)d_in[7];
    const float* g0_b    = (const float*)d_in[8];
    const float* g0_wv   = (const float*)d_in[9];
    const float* L_wh    = (const float*)d_in[10];
    const float* L_ws    = (const float*)d_in[11];
    const float* L_b     = (const float*)d_in[12];
    const float* L_wv    = (const float*)d_in[13];
    float* out = (float*)d_out;

    const int N  = in_sizes[1] / 32;
    const int E  = in_sizes[0] / 2;
    const int Et = E + N;

    // workspace carve-up (floats)
    float* p = (float*)d_ws;
    auto grab = [&](size_t n) { float* r = p; p += n; return r; };
    float* s0    = grab((size_t)N * 32);
    float* v0    = grab((size_t)N * 24);
    float* s_e   = grab((size_t)N * 32);
    float* v_e   = grab((size_t)N * 24);
    float* s_agg = grab((size_t)N * 32);
    float* v_agg = grab((size_t)N * 24);
    float* s_gz  = grab((size_t)N * 32);
    float* v_gz  = grab((size_t)N * 24);
    float* s_gr  = grab((size_t)N * 32);
    float* v_gr  = grab((size_t)N * 24);
    float* s_hh  = grab((size_t)N * 32);
    float* v_hh  = grab((size_t)N * 24);
    float* s_rs  = grab((size_t)N * 32);
    float* v_rs  = grab((size_t)N * 24);
    float* raw_s = grab((size_t)Et);
    float* raw_v = grab((size_t)Et);
    float* exp_s = grab((size_t)Et);
    float* exp_v = grab((size_t)Et);
    unsigned* max_s = (unsigned*)grab((size_t)N);
    unsigned* max_v = (unsigned*)grab((size_t)N);
    float* z_s   = grab((size_t)N);
    float* z_v   = grab((size_t)N);

    const int gvpN  = (N  + 127) / 128;   // 8 wave-tiles of 16 rows per block
    const int gvpE  = (Et + 127) / 128;
    const int thrE  = (Et + 255) / 256;
    const int thrN  = (N  + 255) / 256;

    auto flood = [&](const float* si, const float* vi, int L,
                     float* s_tgt, float* v_tgt, bool accum) {
        const float* wh_e = L_wh + (size_t)(3 * L + 0) * 64;
        const float* ws_e = L_ws + (size_t)(3 * L + 0) * 1280;
        const float* b_e  = L_b  + (size_t)(3 * L + 0) * 32;
        const float* wv_e = L_wv + (size_t)(3 * L + 0) * 64;
        const float* wh_m = L_wh + (size_t)(3 * L + 1) * 64;
        const float* ws_m = L_ws + (size_t)(3 * L + 1) * 1280;
        const float* b_m  = L_b  + (size_t)(3 * L + 1) * 32;
        const float* wv_m = L_wv + (size_t)(3 * L + 1) * 64;
        const float* wh_u = L_wh + (size_t)(3 * L + 2) * 64;
        const float* ws_u = L_ws + (size_t)(3 * L + 2) * 1280;
        const float* b_u  = L_b  + (size_t)(3 * L + 2) * 32;
        const float* wv_u = L_wv + (size_t)(3 * L + 2) * 64;

        // stage a: node GVP (relu, sigmoid)
        k_gvp<8, true, true, false, false><<<gvpN, 256, 0, stream>>>(
            si, vi, wh_e, ws_e, b_e, wv_e, s_e, v_e, N,
            nullptr, 0, 0, nullptr, nullptr, nullptr, nullptr);

        // attention: zero maxes (encoded 0 < encode(-inf)) and z sums
        k_fill<<<thrN, 256, 0, stream>>>((float*)max_s, 0.f, N);
        k_fill<<<thrN, 256, 0, stream>>>((float*)max_v, 0.f, N);
        k_fill<<<thrN, 256, 0, stream>>>(z_s, 0.f, N);
        k_fill<<<thrN, 256, 0, stream>>>(z_v, 0.f, N);
        k_att_raw<<<thrE, 256, 0, stream>>>(s_e, v_e, ei, E, Et,
                                            raw_s, raw_v, max_s, max_v);
        k_att_exp<<<thrE, 256, 0, stream>>>(ei, E, Et, raw_s, raw_v,
                                            max_s, max_v, exp_s, exp_v, z_s, z_v);

        // message GVP + scatter-add aggregation
        k_fill<<<(N * 32 + 255) / 256, 256, 0, stream>>>(s_agg, 0.f, N * 32);
        k_fill<<<(N * 24 + 255) / 256, 256, 0, stream>>>(v_agg, 0.f, N * 24);
        k_gvp<8, true, true, true, false><<<gvpE, 256, 0, stream>>>(
            s_e, v_e, wh_m, ws_m, b_m, wv_m, s_agg, v_agg, Et,
            ei, E, Et, exp_s, exp_v, z_s, z_v);

        // stage f: update GVP into gate accumulator
        if (accum)
            k_gvp<8, true, true, false, true><<<gvpN, 256, 0, stream>>>(
                s_agg, v_agg, wh_u, ws_u, b_u, wv_u, s_tgt, v_tgt, N,
                nullptr, 0, 0, nullptr, nullptr, nullptr, nullptr);
        else
            k_gvp<8, true, true, false, false><<<gvpN, 256, 0, stream>>>(
                s_agg, v_agg, wh_u, ws_u, b_u, wv_u, s_tgt, v_tgt, N,
                nullptr, 0, 0, nullptr, nullptr, nullptr, nullptr);
    };

    // initial GVP (no activations), v has 4 channels
    k_gvp<4, false, false, false, false><<<gvpN, 256, 0, stream>>>(
        s_raw, v_raw, g0_wh, g0_ws, g0_b, g0_wv, s0, v0, N,
        nullptr, 0, 0, nullptr, nullptr, nullptr, nullptr);

    flood(s0, v0, 0, s_gz, v_gz, false);
    flood(s_hid, v_hid, 1, s_gz, v_gz, true);
    flood(s0, v0, 2, s_gr, v_gr, false);
    flood(s_hid, v_hid, 3, s_gr, v_gr, true);
    flood(s0, v0, 4, s_hh, v_hh, false);
    k_gate_scale<<<(N * 32 + 255) / 256, 256, 0, stream>>>(s_gr, s_hid, s_rs, N * 32);
    k_gate_scale<<<(N * 24 + 255) / 256, 256, 0, stream>>>(v_gr, v_hid, v_rs, N * 24);
    flood(s_rs, v_rs, 5, s_hh, v_hh, true);

    k_final<<<(N * 56 + 255) / 256, 256, 0, stream>>>(
        s_gz, s_hh, s_hid, v_gz, v_hh, v_hid, mask, out, N);
}
